// SimpleGNN_37022618091718
// MI455X (gfx1250) — matrix operand order repaired and checked by reading, not compiled, not run on hardware
//
#include <hip/hip_runtime.h>
#include <hip/hip_bf16.h>

typedef __attribute__((ext_vector_type(16))) _Float16 v16h;
typedef __attribute__((ext_vector_type(8)))  _Float16 v8h;
typedef __attribute__((ext_vector_type(8)))  float    v8f;

#define MT 4   // row-tiles (of 16) per wave: 64 rows x 16 cols per wave

// ---------------------------------------------------------------------------
// Degree / normalization kernels
// ---------------------------------------------------------------------------
__global__ void gcn_init_deg(float* __restrict__ deg, int n) {
    int i = blockIdx.x * blockDim.x + threadIdx.x;
    if (i < n) deg[i] = 1.0f;  // self-loop contributes 1
}

__global__ void gcn_accum_deg(const long long* __restrict__ dst,
                              float* __restrict__ deg, long long E) {
    long long e = (long long)blockIdx.x * blockDim.x + threadIdx.x;
    if (e < E) atomicAdd(&deg[dst[e]], 1.0f);
}

__global__ void gcn_finish_dis(float* __restrict__ dis, int n) {
    int i = blockIdx.x * blockDim.x + threadIdx.x;
    if (i < n) dis[i] = rsqrtf(dis[i]);  // deg >= 1 always (self-loops)
}

// ---------------------------------------------------------------------------
// fp32 -> f16 row-major convert, 8 elements per thread (total8 = count/8)
// ---------------------------------------------------------------------------
__global__ void gcn_cvt_f16(const float* __restrict__ s, _Float16* __restrict__ d,
                            long long total8) {
    long long i = (long long)blockIdx.x * blockDim.x + threadIdx.x;
    if (i < total8) {
        const float4* s4 = (const float4*)s + i * 2;
        float4 u = s4[0], v = s4[1];
        v8h o;
        o[0] = (_Float16)u.x; o[1] = (_Float16)u.y;
        o[2] = (_Float16)u.z; o[3] = (_Float16)u.w;
        o[4] = (_Float16)v.x; o[5] = (_Float16)v.y;
        o[6] = (_Float16)v.z; o[7] = (_Float16)v.w;
        *((v8h*)d + i) = o;
    }
}

// ---------------------------------------------------------------------------
// Pack B[K,N] (fp32 row-major) into fragment-major f16:
//   P[((nt*ksteps + ks)*32 + lane)*16 + i] = B[ks*32 + (lane<16?0:16) + i][nt*16 + lane%16]
// One thread per (nt, ks, lane).
// ---------------------------------------------------------------------------
__global__ void gcn_pack_b(const float* __restrict__ B, _Float16* __restrict__ P,
                           int K, int N) {
    int ksteps = K >> 5;
    int total  = (N >> 4) * ksteps * 32;
    int tid = blockIdx.x * blockDim.x + threadIdx.x;
    if (tid >= total) return;
    int lane = tid & 31;
    int ks   = (tid >> 5) % ksteps;
    int nt   = tid / (32 * ksteps);
    int l15  = lane & 15;
    int kb   = ks * 32 + (lane >> 4) * 16;
    int n    = nt * 16 + l15;
    _Float16* o = P + (size_t)tid * 16;
    #pragma unroll
    for (int i = 0; i < 16; ++i)
        o[i] = (_Float16)B[(size_t)(kb + i) * N + n];
}

// ---------------------------------------------------------------------------
// WMMA GEMM: C[M,N] = A16[M,K] @ B (packed f16), fp32 accumulate/out.
// One wave per 64x16 output block (4 row-tiles share one B fragment).
// A16: f16 row-major. Bp: fragment-major from gcn_pack_b. K mult of 32.
// ---------------------------------------------------------------------------
__global__ __launch_bounds__(32)
void gcn_wmma_gemm(const _Float16* __restrict__ A, const _Float16* __restrict__ Bp,
                   float* __restrict__ C, int M, int K, int N) {
    const int lane = threadIdx.x;        // 0..31 (wave32)
    const int half = lane >> 4;          // 0: lanes 0-15, 1: lanes 16-31
    const int l15  = lane & 15;
    const int tn   = blockIdx.x * 16;          // col-tile base
    const int tm0  = blockIdx.y * (16 * MT);   // first row of 64-row block

    const int n = tn + l15;              // B/C column handled by this lane
    const int ksteps = K >> 5;

    // A-row per tile, clamped so all lanes stay active (WMMA needs full EXEC)
    const _Float16* arow[MT];
    #pragma unroll
    for (int t = 0; t < MT; ++t) {
        int m = tm0 + 16 * t + l15;
        if (m > M - 1) m = M - 1;
        arow[t] = A + (size_t)m * K;
    }

    const int kbaseA = half * 16;        // A: lanes16-31 hold K=16.. (mod 32)
    // B fragments for this col-tile, lane-contiguous 32B each
    const v16h* bp = (const v16h*)Bp + (size_t)blockIdx.x * ksteps * 32 + lane;

    v8f c[MT];
    #pragma unroll
    for (int t = 0; t < MT; ++t) c[t] = (v8f){};

    for (int ks = 0; ks < ksteps; ++ks) {
        v16h b = bp[(size_t)ks * 32];
        #pragma unroll
        for (int t = 0; t < MT; ++t) {
            // A fragment: halves 0-7 -> K=kbaseA+0..7 ; 8-15 -> kbaseA+8..15
            const _Float16* ap = arow[t] + ks * 32 + kbaseA;
            v8h lo = *(const v8h*)ap;            // 16B aligned
            v8h hi = *(const v8h*)(ap + 8);      // 16B aligned
            v16h a = __builtin_shufflevector(lo, hi,
                         0,1,2,3,4,5,6,7,8,9,10,11,12,13,14,15);
            c[t] = __builtin_amdgcn_wmma_f32_16x16x32_f16(
                       /*neg_a=*/false, a, /*neg_b=*/false, b,
                       /*c_mod=*/(short)0, c[t], /*reuse_a=*/false, /*reuse_b=*/false);
        }
    }

    // C/D layout: VGPR r -> row (tile_base + half*8 + r), column n
    #pragma unroll
    for (int t = 0; t < MT; ++t) {
        const int row0 = tm0 + 16 * t + half * 8;
        float* cp = C + (size_t)row0 * N + n;
        if (row0 + 7 < M) {              // fast path: whole half-tile in range
            #pragma unroll
            for (int r = 0; r < 8; ++r) { *cp = c[t][r]; cp += N; }
        } else {
            #pragma unroll
            for (int r = 0; r < 8; ++r) {
                if (row0 + r < M) *cp = c[t][r];
                cp += N;
            }
        }
    }
}

// ---------------------------------------------------------------------------
// Aggregation: acc[i] = dis[i]^2 * xw[i]  (self-loop term, also inits acc)
// ---------------------------------------------------------------------------
__global__ void gcn_agg_self(const float* __restrict__ xw,
                             const float* __restrict__ dis,
                             float* __restrict__ acc, long long total, int fshift) {
    long long idx = (long long)blockIdx.x * blockDim.x + threadIdx.x;
    if (idx < total) {
        float d = dis[idx >> fshift];
        acc[idx] = d * d * xw[idx];
    }
}

// ---------------------------------------------------------------------------
// Aggregation over edges: one wave per edge; lanes cover features (float4).
// acc[dst] += dis[src]*dis[dst] * xw[src]
// ---------------------------------------------------------------------------
__global__ __launch_bounds__(256)
void gcn_agg_edges(const long long* __restrict__ src,
                   const long long* __restrict__ dst,
                   const float* __restrict__ dis,
                   const float* __restrict__ xw,
                   float* __restrict__ acc, long long E, int F) {
    long long warp = ((long long)blockIdx.x * blockDim.x + threadIdx.x) >> 5;
    int lane = threadIdx.x & 31;
    if (warp >= E) return;
    long long s = src[warp];
    long long d = dst[warp];
    float w = dis[s] * dis[d];
    const float* xr = xw  + (size_t)s * F;
    float*       ar = acc + (size_t)d * F;
    for (int f = lane * 4; f < F; f += 128) {  // F is a multiple of 128
        float4 v = *(const float4*)(xr + f);
        atomicAdd(ar + f + 0, w * v.x);
        atomicAdd(ar + f + 1, w * v.y);
        atomicAdd(ar + f + 2, w * v.z);
        atomicAdd(ar + f + 3, w * v.w);
    }
}

// ---------------------------------------------------------------------------
// In-place bias (+ optional ReLU)
// ---------------------------------------------------------------------------
__global__ void gcn_bias_act(float* __restrict__ acc,
                             const float* __restrict__ bias,
                             long long total, int fmask, int do_relu) {
    long long idx = (long long)blockIdx.x * blockDim.x + threadIdx.x;
    if (idx < total) {
        float v = acc[idx] + bias[(int)idx & fmask];
        acc[idx] = do_relu ? fmaxf(v, 0.0f) : v;
    }
}

// ---------------------------------------------------------------------------
extern "C" void kernel_launch(void* const* d_in, const int* in_sizes, int n_in,
                              void* d_out, int out_size, void* d_ws, size_t ws_size,
                              hipStream_t stream) {
    const float*     x   = (const float*)d_in[0];
    const long long* ei  = (const long long*)d_in[1];   // int64 [2, E]
    const float*     W1  = (const float*)d_in[2];
    const float*     b1  = (const float*)d_in[3];
    const float*     W2  = (const float*)d_in[4];
    const float*     b2  = (const float*)d_in[5];
    float*           out = (float*)d_out;

    const long long E   = (long long)in_sizes[1] / 2;
    const int HID = in_sizes[3];                 // 256
    const int OUT = in_sizes[5];                 // 128
    const int IN  = in_sizes[2] / HID;           // 128
    const int N   = in_sizes[0] / IN;            // 100000

    const long long* src = ei;
    const long long* dst = ei + E;

    // Workspace (float units):
    //   dis[N] | xw[N*HID] (xw1, reused as hw2) | hac[N*HID] |
    //   a16 (N*HID halves = N*HID/2 floats) | bpack (IN*HID halves max)
    float* ws  = (float*)d_ws;
    float* dis = ws;
    size_t off = ((size_t)N + 255) & ~(size_t)255;
    float* xw  = ws + off;
    float* hac = xw + (size_t)N * HID;
    _Float16* a16   = (_Float16*)(hac + (size_t)N * HID);
    _Float16* bpack = a16 + (size_t)N * HID;

    const int TB = 256;
    long long tot1 = (long long)N * HID;
    long long tot2 = (long long)N * OUT;
    long long ain8_1 = (long long)N * IN  / 8;   // x  -> f16, 8 elems/thread
    long long ain8_2 = (long long)N * HID / 8;   // h  -> f16
    int hshift = 31 - __builtin_clz((unsigned)HID);
    int oshift = 31 - __builtin_clz((unsigned)OUT);

    // 1) symmetric-normalization degrees
    gcn_init_deg  <<<(N + TB - 1) / TB, TB, 0, stream>>>(dis, N);
    gcn_accum_deg <<<(unsigned)((E + TB - 1) / TB), TB, 0, stream>>>(dst, dis, E);
    gcn_finish_dis<<<(N + TB - 1) / TB, TB, 0, stream>>>(dis, N);

    // 2) xw = x @ W1   (f16 WMMA with pre-converted operands)
    gcn_cvt_f16<<<(unsigned)((ain8_1 + TB - 1) / TB), TB, 0, stream>>>(x, a16, ain8_1);
    {
        int packthreads = (HID / 16) * (IN / 32) * 32;
        gcn_pack_b<<<(packthreads + TB - 1) / TB, TB, 0, stream>>>(W1, bpack, IN, HID);
    }
    dim3 g1(HID / 16, (N + 16 * MT - 1) / (16 * MT));
    gcn_wmma_gemm<<<g1, 32, 0, stream>>>(a16, bpack, xw, N, IN, HID);

    // 3) layer-1 aggregation + bias + ReLU  -> hac
    gcn_agg_self <<<(unsigned)((tot1 + TB - 1) / TB), TB, 0, stream>>>(xw, dis, hac, tot1, hshift);
    gcn_agg_edges<<<(unsigned)((E * 32 + TB - 1) / TB), TB, 0, stream>>>(src, dst, dis, xw, hac, E, HID);
    gcn_bias_act <<<(unsigned)((tot1 + TB - 1) / TB), TB, 0, stream>>>(hac, b1, tot1, HID - 1, 1);

    // 4) hw2 = h @ W2  (reuse xw buffer for output)
    gcn_cvt_f16<<<(unsigned)((ain8_2 + TB - 1) / TB), TB, 0, stream>>>(hac, a16, ain8_2);
    {
        int packthreads = (OUT / 16) * (HID / 32) * 32;
        gcn_pack_b<<<(packthreads + TB - 1) / TB, TB, 0, stream>>>(W2, bpack, HID, OUT);
    }
    dim3 g2(OUT / 16, (N + 16 * MT - 1) / (16 * MT));
    gcn_wmma_gemm<<<g2, 32, 0, stream>>>(a16, bpack, xw, N, HID, OUT);

    // 5) layer-2 aggregation + bias -> out
    gcn_agg_self <<<(unsigned)((tot2 + TB - 1) / TB), TB, 0, stream>>>(xw, dis, out, tot2, oshift);
    gcn_agg_edges<<<(unsigned)((E * 32 + TB - 1) / TB), TB, 0, stream>>>(src, dst, dis, xw, out, E, OUT);
    gcn_bias_act <<<(unsigned)((tot2 + TB - 1) / TB), TB, 0, stream>>>(out, b2, tot2, OUT - 1, 0);
}
